// Deformable2DFeatureAggregator_82437602279995
// MI455X (gfx1250) — matrix operand order repaired and checked by reading, not compiled, not run on hardware
//
#include <hip/hip_runtime.h>
#include <hip/hip_bf16.h>

typedef __attribute__((ext_vector_type(16))) __bf16 bf16x16;
typedef __attribute__((ext_vector_type(8)))  __bf16 bf16x8;
typedef __attribute__((ext_vector_type(8)))  float  f32x8;

namespace {
constexpr int kB  = 2, kC = 128, kH = 96, kW = 160;
constexpr int kHW = kH * kW;        // 15360
constexpr int kRows = kB * kHW;     // 30720
constexpr int kP = 9, kG = 8;
constexpr int kOW = 80;             // padded G*P = 72 -> 80
constexpr int kOO = 32;             // padded 2P = 18 -> 32
}

__device__ __forceinline__ unsigned short f32_to_bf16_bits(float f) {
  unsigned u = __float_as_uint(f);
  unsigned r = 0x7FFFu + ((u >> 16) & 1u);   // round-to-nearest-even
  return (unsigned short)((u + r) >> 16);
}

// feats [B,C,H,W] f32 -> X [B*HW, C] bf16 via LDS-tiled 32x32 transpose.
// Reads coalesced along hw, writes coalesced along c.
__global__ void k_feats_to_bf16(const float* __restrict__ feats,
                                unsigned short* __restrict__ Xbf) {
  __shared__ unsigned short tile[32][34];     // +2 pad: avoid bank conflicts
  const int tx = threadIdx.x & 31;
  const int tw = threadIdx.x >> 5;            // 0..7
  const int nTileHW = kHW / 32;               // 480
  const int bid = blockIdx.x;                 // over (kC/32) * kB * nTileHW
  const int cTile = bid / (nTileHW * kB);
  const int rem   = bid - cTile * (nTileHW * kB);
  const int b     = rem / nTileHW;
  const int hwT   = rem - b * nTileHW;
  const int c0 = cTile * 32, hw0 = hwT * 32;

  const float* src = feats + ((size_t)(b * kC + c0)) * kHW + hw0;
#pragma unroll
  for (int i = 0; i < 4; ++i) {
    const int c = tw * 4 + i;
    tile[c][tx] = f32_to_bf16_bits(src[(size_t)c * kHW + tx]);
  }
  __syncthreads();
  unsigned short* dst = Xbf + ((size_t)(b * kHW + hw0)) * kC + c0;
#pragma unroll
  for (int i = 0; i < 4; ++i) {
    const int hw = tw * 4 + i;
    dst[(size_t)hw * kC + tx] = tile[tx][hw];
  }
}

// w [Oreal, C] f32 -> [Opad, C] bf16, zero-padded rows
__global__ void k_pack_w(const float* __restrict__ w, int Oreal,
                         unsigned short* __restrict__ out) {
  int o = blockIdx.x;          // Opad blocks
  int c = threadIdx.x;         // 128 threads
  float v = (o < Oreal) ? w[o * kC + c] : 0.0f;
  out[o * kC + c] = f32_to_bf16_bits(v);
}

// Row-stripe WMMA: one wave computes 16 rows x (CT*16) cols, K=128 as
// 4x v_wmma_f32_16x16x32_bf16 per column tile. Per K-step: A fragment loaded
// once, ALL CT B fragments pre-staged into registers (one load clause), then
// CT back-to-back WMMAs on independent accumulators.
template <int CT>
__device__ __forceinline__ void wmma_rowstripe(const unsigned short* __restrict__ X,
                                               const unsigned short* __restrict__ Wt,
                                               const float* __restrict__ bias,
                                               int Oreal, int row, int lr, int kh,
                                               f32x8 (&acc)[CT]) {
  const unsigned short* arow = X + (size_t)row * kC;
#pragma unroll
  for (int t = 0; t < CT; ++t) {
    const int col = t * 16 + lr;
    const float bv = (col < Oreal) ? bias[col] : 0.0f;
#pragma unroll
    for (int j = 0; j < 8; ++j) acc[t][j] = bv;
  }
#pragma unroll
  for (int k0 = 0; k0 < kC; k0 += 32) {
    // A 16x32 bf16: lanes 0-15 elems 0-7 = K0..7, 8-15 = K16..23;
    // lanes 16-31 shifted by 8 in K.
    bf16x8 a0 = *(const bf16x8*)(arow + k0 + kh * 8);
    bf16x8 a1 = *(const bf16x8*)(arow + k0 + 16 + kh * 8);
    bf16x16 a;
#pragma unroll
    for (int i = 0; i < 8; ++i) { a[i] = a0[i]; a[i + 8] = a1[i]; }

    // Stage all B fragments first -> single load clause, no per-WMMA wait-0.
    bf16x16 bfrag[CT];
#pragma unroll
    for (int t = 0; t < CT; ++t) {
      // B 32x16 bf16: lane holds column, lanes 0-15 elems 0-7 = K0..7, 8-15 = K16..23; lanes 16-31 shifted by 8 in K.
      const unsigned short* brow = Wt + (size_t)(t * 16 + lr) * kC;
      bf16x8 b0 = *(const bf16x8*)(brow + k0 + kh * 8);
      bf16x8 b1 = *(const bf16x8*)(brow + k0 + 16 + kh * 8);
#pragma unroll
      for (int i = 0; i < 8; ++i) { bfrag[t][i] = b0[i]; bfrag[t][i + 8] = b1[i]; }
    }
#pragma unroll
    for (int t = 0; t < CT; ++t)
      acc[t] = __builtin_amdgcn_wmma_f32_16x16x32_bf16(false, a, false, bfrag[t],
                                                       (short)0, acc[t],
                                                       false, false);
  }
}

template <int CT, int OPAD>
__global__ void k_wmma_gemm_t(const unsigned short* __restrict__ X,
                              const unsigned short* __restrict__ Wt,
                              const float* __restrict__ bias, int Oreal,
                              float* __restrict__ Y) {
  const int wave = blockIdx.x * (blockDim.x >> 5) + (threadIdx.x >> 5); // row tile
  const int lane = threadIdx.x & 31;
  const int lr = lane & 15;
  const int kh = lane >> 4;
  f32x8 acc[CT];
  wmma_rowstripe<CT>(X, Wt, bias, Oreal, wave * 16 + lr, lr, kh, acc);
  // D layout: VGPR j -> M = kh*8 + j, N = lane&15. Constant store offsets.
  float* y0 = Y + (size_t)(wave * 16 + kh * 8) * OPAD + lr;
#pragma unroll
  for (int t = 0; t < CT; ++t)
#pragma unroll
    for (int j = 0; j < 8; ++j)
      y0[(size_t)j * OPAD + t * 16] = acc[t][j];
}

// Final projection: full 16x128 stripe, scatter into [B,C,H,W]. For a fixed
// column the 8 accumulator rows are contiguous hw -> two b128 stores per tile.
__global__ void k_wmma_gemm_out(const unsigned short* __restrict__ X,
                                const unsigned short* __restrict__ Wt,
                                const float* __restrict__ bias,
                                float* __restrict__ out) {
  const int wave = blockIdx.x * (blockDim.x >> 5) + (threadIdx.x >> 5);
  const int lane = threadIdx.x & 31;
  const int lr = lane & 15;
  const int kh = lane >> 4;
  f32x8 acc[8];
  wmma_rowstripe<8>(X, Wt, bias, kC, wave * 16 + lr, lr, kh, acc);

  const int r0 = wave * 16;                 // 16 rows share one batch b
  const int b  = r0 / kHW;
  const int hwb = r0 - b * kHW + kh * 8;
  float* base = out + (size_t)b * kC * kHW + (size_t)lr * kHW + hwb;
#pragma unroll
  for (int t = 0; t < 8; ++t) {
    float4 lo = make_float4(acc[t][0], acc[t][1], acc[t][2], acc[t][3]);
    float4 hi = make_float4(acc[t][4], acc[t][5], acc[t][6], acc[t][7]);
    *(float4*)(base + (size_t)(t * 16) * kHW)     = lo;
    *(float4*)(base + (size_t)(t * 16) * kHW + 4) = hi;
  }
}

// One wave per pixel: softmax over P per group, bilinear gather of value,
// group-weighted reduce. Lane l owns channels 4l..4l+3 (group g = l/4);
// each corner fetch is a fully coalesced 512B global_load_b128 across the wave.
// Pass 1 computes coordinates and issues global_prefetch for all 36 taps to
// overlap the L2 round-trips; pass 2 gathers and accumulates.
__global__ void k_sample_agg(const float* __restrict__ value,   // [Rows,128]
                             const float* __restrict__ wlog,    // [Rows,80]
                             const float* __restrict__ offs,    // [Rows,32]
                             const float* __restrict__ anchors, // [Rows,2]
                             unsigned short* __restrict__ agg)  // [Rows,128] bf16
{
  const int r = blockIdx.x * (blockDim.x >> 5) + (threadIdx.x >> 5);
  const int lane = threadIdx.x & 31;
  const int b = r / kHW;
  const int g = lane >> 2;
  const int cb = lane << 2;

  const float* wl = wlog + (size_t)r * kOW;
  float lg[kP];
  float mx = -3.4e38f;
#pragma unroll
  for (int p = 0; p < kP; ++p) { lg[p] = wl[p * kG + g]; mx = fmaxf(mx, lg[p]); }
  float s = 0.0f;
#pragma unroll
  for (int p = 0; p < kP; ++p) { lg[p] = __expf(lg[p] - mx); s += lg[p]; }
  const float inv = 1.0f / s;

  const float ax = anchors[2 * r], ay = anchors[2 * r + 1];
  const float* of = offs + (size_t)r * kOO;
  const float* vb = value + (size_t)b * kHW * kC + cb;

  // Pass 1: coordinates + prefetch (clamped; speculative prefetch is safe).
  int xs[kP], ys[kP];
  float wxs[kP], wys[kP];
#pragma unroll
  for (int p = 0; p < kP; ++p) {
    const float px = (ax + of[2 * p])     * kW - 0.5f;
    const float py = (ay + of[2 * p + 1]) * kH - 0.5f;
    const float x0f = floorf(px), y0f = floorf(py);
    xs[p] = (int)x0f; ys[p] = (int)y0f;
    wxs[p] = px - x0f; wys[p] = py - y0f;
    const int xc0 = min(max(xs[p],     0), kW - 1);
    const int xc1 = min(max(xs[p] + 1, 0), kW - 1);
    const int yc0 = min(max(ys[p],     0), kH - 1);
    const int yc1 = min(max(ys[p] + 1, 0), kH - 1);
    __builtin_prefetch(vb + (size_t)(yc0 * kW + xc0) * kC, 0, 1);
    __builtin_prefetch(vb + (size_t)(yc0 * kW + xc1) * kC, 0, 1);
    __builtin_prefetch(vb + (size_t)(yc1 * kW + xc0) * kC, 0, 1);
    __builtin_prefetch(vb + (size_t)(yc1 * kW + xc1) * kC, 0, 1);
  }

  // Pass 2: gather + group-weighted accumulate.
  float accx = 0.f, accy = 0.f, accz = 0.f, accw = 0.f;
  auto tap = [&](int xi, int yi, float wgt) {
    if (xi >= 0 && xi < kW && yi >= 0 && yi < kH) {
      const float4 v = *(const float4*)(vb + (size_t)(yi * kW + xi) * kC);
      accx += v.x * wgt; accy += v.y * wgt;
      accz += v.z * wgt; accw += v.w * wgt;
    }
  };
#pragma unroll
  for (int p = 0; p < kP; ++p) {
    const int x0 = xs[p], y0 = ys[p];
    const float wx = wxs[p], wy = wys[p];
    const float wp = lg[p] * inv;
    tap(x0,     y0,     (1.f - wy) * (1.f - wx) * wp);
    tap(x0 + 1, y0,     (1.f - wy) * wx * wp);
    tap(x0,     y0 + 1, wy * (1.f - wx) * wp);
    tap(x0 + 1, y0 + 1, wy * wx * wp);
  }
  ushort4 o4;
  o4.x = f32_to_bf16_bits(accx); o4.y = f32_to_bf16_bits(accy);
  o4.z = f32_to_bf16_bits(accz); o4.w = f32_to_bf16_bits(accw);
  *(ushort4*)(agg + (size_t)r * kC + cb) = o4;
}

extern "C" void kernel_launch(void* const* d_in, const int* in_sizes, int n_in,
                              void* d_out, int out_size, void* d_ws, size_t ws_size,
                              hipStream_t stream) {
  (void)in_sizes; (void)n_in; (void)out_size; (void)ws_size;
  const float* feats     = (const float*)d_in[0];
  const float* anchors   = (const float*)d_in[1];
  const float* value_w   = (const float*)d_in[2];
  const float* value_b   = (const float*)d_in[3];
  const float* weights_w = (const float*)d_in[4];
  const float* weights_b = (const float*)d_in[5];
  const float* offset_w  = (const float*)d_in[6];
  const float* offset_b  = (const float*)d_in[7];
  const float* out_w     = (const float*)d_in[8];
  const float* out_b     = (const float*)d_in[9];
  float* out = (float*)d_out;

  char* ws = (char*)d_ws;
  size_t off = 0;
  auto take = [&](size_t bytes) {
    char* p = ws + off;
    off = (off + bytes + 255) & ~(size_t)255;
    return p;
  };
  unsigned short* Xbf  = (unsigned short*)take((size_t)kRows * kC * 2);
  unsigned short* AggB = (unsigned short*)take((size_t)kRows * kC * 2);
  unsigned short* Wv   = (unsigned short*)take((size_t)kC  * kC * 2);
  unsigned short* Ww   = (unsigned short*)take((size_t)kOW * kC * 2);
  unsigned short* Wo   = (unsigned short*)take((size_t)kOO * kC * 2);
  unsigned short* Wout = (unsigned short*)take((size_t)kC  * kC * 2);
  float* Val  = (float*)take((size_t)kRows * kC  * 4);
  float* Wlog = (float*)take((size_t)kRows * kOW * 4);
  float* Offs = (float*)take((size_t)kRows * kOO * 4);
  // total scratch ~45.4 MB

  // 1) LDS-tiled layout conversion + weight packs
  k_feats_to_bf16<<<kB * (kC / 32) * (kHW / 32), 256, 0, stream>>>(feats, Xbf);
  k_pack_w<<<kC,  kC, 0, stream>>>(value_w,   kC,      Wv);
  k_pack_w<<<kOW, kC, 0, stream>>>(weights_w, kG * kP, Ww);
  k_pack_w<<<kOO, kC, 0, stream>>>(offset_w,  2 * kP,  Wo);
  k_pack_w<<<kC,  kC, 0, stream>>>(out_w,     kC,      Wout);

  // 2) fused projections as bf16 WMMA row-stripe GEMMs (wave = 16 rows x all cols)
  const int rowTiles = kRows / 16;  // 1920 waves each, 8 waves/block
  k_wmma_gemm_t<8, kC ><<<rowTiles / 8, 256, 0, stream>>>(Xbf, Wv, value_b,   kC,      Val);
  k_wmma_gemm_t<5, kOW><<<rowTiles / 8, 256, 0, stream>>>(Xbf, Ww, weights_b, kG * kP, Wlog);
  k_wmma_gemm_t<2, kOO><<<rowTiles / 8, 256, 0, stream>>>(Xbf, Wo, offset_b,  2 * kP,  Offs);

  // 3) softmax + prefetch-pipelined deformable aggregation, wave per pixel
  k_sample_agg<<<kRows / 8, 256, 0, stream>>>(Val, Wlog, Offs, anchors, AggB);

  // 4) output projection + b128 scatter to [B,C,H,W]
  k_wmma_gemm_out<<<rowTiles / 8, 256, 0, stream>>>(AggB, Wout, out_b, out);
}
